// Generator_31164282700384
// MI455X (gfx1250) — matrix operand order repaired and checked by reading, not compiled, not run on hardware
//
#include <hip/hip_runtime.h>

// Problem dims
#define Bn 4096
#define Hn 512
#define Fn 256
#define Zn 64
#define Tn 16
// LSTM GEMM K = (Z + H + F) + H = 832 + 512 = 1344

typedef __attribute__((ext_vector_type(16))) __bf16 v16bf;
typedef __attribute__((ext_vector_type(8)))  __bf16 v8bf;
typedef __attribute__((ext_vector_type(8)))  float  v8f;

__device__ __forceinline__ unsigned short f2bf(float f) {
  unsigned u = __builtin_bit_cast(unsigned, f);
  return (unsigned short)((u + 0x7FFFu + ((u >> 16) & 1u)) >> 16); // RNE
}

// ---------------------------------------------------------------------------
// Weight pack: fp32 row-major W[K,N] -> bf16 WMMA-B fragment tiles.
// Tile (kt,nt): 32 lanes x 16 bf16 (1KB). Lane L covers col n = nt*16+(L&15),
// element i covers k = kt*32 + (L>>4)*16 + i.  GEMM loads one contiguous
// v16bf per lane per fragment.  Packed offset: ((nt*KTtot + kt)*32 + L)*16.
// ---------------------------------------------------------------------------
__global__ __launch_bounds__(32)
void pack_w_kernel(const float* __restrict__ W, unsigned short* __restrict__ P,
                   int N, int KTtot, int ktOff) {
  const int kt = blockIdx.x, nt = blockIdx.y, L = threadIdx.x;
  const int n  = nt * 16 + (L & 15);
  const int kb = kt * 32 + (L >> 4) * 16;
  unsigned short* dst = P + (((size_t)nt * KTtot + ktOff + kt) * 32 + L) * 16;
#pragma unroll
  for (int i = 0; i < 16; ++i)
    dst[i] = f2bf(W[(size_t)(kb + i) * N + n]);
}

// ---------------------------------------------------------------------------
// One K-segment of the wave-level GEMM.  Direct kernel-arg pointer use so the
// compiler promotes A loads to global_load (no flat path / DScnt coupling).
// ---------------------------------------------------------------------------
__device__ __forceinline__ void gemm_seg(
    const unsigned short* __restrict__ A, int K, int gk0,
    const unsigned short* __restrict__ P, int KTtot,
    int m0, int nt0, int lane, v8f acc[4][4]) {
  const int lm = lane & 15, lh = lane >> 4;
  for (int kk = 0; kk < K; kk += 32) {
    // A fragments: lane = row (lm), lane-half selects contiguous K groups:
    // lanes 0-15: K {0..15}; lanes 16-31: K {16..31}
    v16bf af[4];
#pragma unroll
    for (int ti = 0; ti < 4; ++ti) {
      const unsigned short* ap =
          A + (size_t)(m0 + ti * 16 + lm) * K + kk + lh * 16;
      v8bf lo = *(const v8bf*)ap;
      v8bf hi = *(const v8bf*)(ap + 8);
      af[ti] = __builtin_shufflevector(lo, hi, 0, 1, 2, 3, 4, 5, 6, 7, 8, 9,
                                       10, 11, 12, 13, 14, 15);
    }
    // B fragments: one contiguous 32B load per lane from packed tiles
    const int gkt = (gk0 + kk) >> 5;
    const unsigned short* bp =
        P + ((size_t)nt0 * KTtot + gkt) * 512 + lane * 16;
    v16bf bfrag[4];
#pragma unroll
    for (int tj = 0; tj < 4; ++tj)
      bfrag[tj] = *(const v16bf*)(bp + (size_t)tj * KTtot * 512);

#pragma unroll
    for (int ti = 0; ti < 4; ++ti)
#pragma unroll
      for (int tj = 0; tj < 4; ++tj)
        acc[ti][tj] = __builtin_amdgcn_wmma_f32_16x16x32_bf16(
            false, af[ti], false, bfrag[tj], (short)0, acc[ti][tj], false,
            false);
  }
}

// ---------------------------------------------------------------------------
// Segmented-A bf16 WMMA GEMM.  One wave computes a 64x64 fp32 tile.
// A: up to 4 row-major bf16 segments [M, k_s] (K concat = sum k_s, each %32==0)
// P: packed weights, KTtot = total K tiles.  Epilogue: + bias, compile-time
// ReLU / fp32 store (ldcf) / bf16 store (ldcb).
// ---------------------------------------------------------------------------
template <bool RELU, bool HASF, bool HASB>
__global__ __launch_bounds__(32)
void gemm_wmma_bf16(const unsigned short* __restrict__ a0, int k0,
                    const unsigned short* __restrict__ a1, int k1,
                    const unsigned short* __restrict__ a2, int k2,
                    const unsigned short* __restrict__ a3, int k3,
                    const unsigned short* __restrict__ P, int KTtot,
                    const float* __restrict__ bias,
                    float* __restrict__ Cf, int ldcf,
                    unsigned short* __restrict__ Cb, int ldcb) {
  const int lane = threadIdx.x;
  const int lm = lane & 15, lh = lane >> 4;
  const int m0  = blockIdx.y * 64;
  const int nt0 = blockIdx.x * 4;   // base N-tile (16 cols each)

  v8f acc[4][4] = {};

  int gk = 0;
  if (k0) { gemm_seg(a0, k0, gk, P, KTtot, m0, nt0, lane, acc); gk += k0; }
  if (k1) { gemm_seg(a1, k1, gk, P, KTtot, m0, nt0, lane, acc); gk += k1; }
  if (k2) { gemm_seg(a2, k2, gk, P, KTtot, m0, nt0, lane, acc); gk += k2; }
  if (k3) { gemm_seg(a3, k3, gk, P, KTtot, m0, nt0, lane, acc); gk += k3; }

  float bv[4];
#pragma unroll
  for (int tj = 0; tj < 4; ++tj)
    bv[tj] = bias[nt0 * 16 + tj * 16 + lm];

  // C/D layout: VGPR r, lanes 0-15 -> M = r, lanes 16-31 -> M = 8 + r; N = lm
#pragma unroll
  for (int ti = 0; ti < 4; ++ti) {
    const int rbase = m0 + ti * 16 + lh * 8;
#pragma unroll
    for (int tj = 0; tj < 4; ++tj) {
      const int col = nt0 * 16 + tj * 16 + lm;
#pragma unroll
      for (int r = 0; r < 8; ++r) {
        float v = acc[ti][tj][r] + bv[tj];
        if (RELU) v = fmaxf(v, 0.0f);
        if (HASF) Cf[(size_t)(rbase + r) * ldcf + col] = v;
        if (HASB) Cb[(size_t)(rbase + r) * ldcb + col] = f2bf(v);
      }
    }
  }
}

// ---------------------------------------------------------------------------
// LSTM cell pointwise: gates [B,4H] (i,f,g,o), c updated in place (fp32),
// bf16 copies of c_new and h_new for the following GEMMs.
// ---------------------------------------------------------------------------
__global__ void lstm_pointwise_kernel(const float* __restrict__ gates,
                                      float* __restrict__ c,
                                      unsigned short* __restrict__ c_bf,
                                      unsigned short* __restrict__ h_bf) {
  int idx = blockIdx.x * blockDim.x + threadIdx.x;
  if (idx >= Bn * Hn) return;
  int b = idx >> 9, j = idx & (Hn - 1);
  const float* g = gates + (size_t)b * (4 * Hn);
  float ig = g[j], fg = g[Hn + j], gg = g[2 * Hn + j], og = g[3 * Hn + j];
  float si = 1.f / (1.f + __expf(-ig));
  float sf = 1.f / (1.f + __expf(-fg));
  float so = 1.f / (1.f + __expf(-og));
  float cn = sf * c[idx] + si * tanhf(gg);
  float hn = so * tanhf(cn);
  c[idx] = cn;
  c_bf[idx] = f2bf(cn);
  h_bf[idx] = f2bf(hn);
}

// z = mean + eps * exp(0.5*logvar); optional bf16 copy for the recurrence
__global__ void z_sample_kernel(const float* __restrict__ mean_t, int ldm,
                                const float* __restrict__ logv_t, int ldl,
                                const float* __restrict__ eps_t,
                                float* __restrict__ z_out,
                                unsigned short* z_bf) {
  int idx = blockIdx.x * blockDim.x + threadIdx.x;
  if (idx >= Bn * Zn) return;
  int b = idx >> 6, z = idx & 63;
  float m = mean_t[(size_t)b * ldm + z];
  float lv = logv_t[(size_t)b * ldl + z];
  float zv = m + eps_t[idx] * __expf(0.5f * lv);
  z_out[idx] = zv;
  if (z_bf) z_bf[idx] = f2bf(zv);
}

__global__ void cvt_bf16_kernel(const float* __restrict__ s,
                                unsigned short* __restrict__ d, int n) {
  int i = blockIdx.x * blockDim.x + threadIdx.x;
  if (i < n) d[i] = f2bf(s[i]);
}

__global__ void zero_kernel(unsigned int* p, int n) {
  int i = blockIdx.x * blockDim.x + threadIdx.x;
  if (i < n) p[i] = 0u;
}

// ---------------------------------------------------------------------------
static inline char* carve(char*& p, size_t bytes) {
  char* r = p;
  p += (bytes + 255) & ~(size_t)255;
  return r;
}

static inline void launch_gemm(hipStream_t stream, int N,
                               const unsigned short* a0, int k0,
                               const unsigned short* a1, int k1,
                               const unsigned short* a2, int k2,
                               const unsigned short* a3, int k3,
                               const unsigned short* P, int KT,
                               const float* bias, int relu,
                               float* Cf, int ldcf,
                               unsigned short* Cb, int ldcb) {
  dim3 g(N / 64, Bn / 64), blk(32);
  if (!relu) {
    gemm_wmma_bf16<false, true, false><<<g, blk, 0, stream>>>(
        a0, k0, a1, k1, a2, k2, a3, k3, P, KT, bias, Cf, ldcf, Cb, ldcb);
  } else if (Cf && Cb) {
    gemm_wmma_bf16<true, true, true><<<g, blk, 0, stream>>>(
        a0, k0, a1, k1, a2, k2, a3, k3, P, KT, bias, Cf, ldcf, Cb, ldcb);
  } else if (Cf) {
    gemm_wmma_bf16<true, true, false><<<g, blk, 0, stream>>>(
        a0, k0, a1, k1, a2, k2, a3, k3, P, KT, bias, Cf, ldcf, Cb, ldcb);
  } else {
    gemm_wmma_bf16<true, false, true><<<g, blk, 0, stream>>>(
        a0, k0, a1, k1, a2, k2, a3, k3, P, KT, bias, Cf, ldcf, Cb, ldcb);
  }
}

extern "C" void kernel_launch(void* const* d_in, const int* in_sizes, int n_in,
                              void* d_out, int out_size, void* d_ws,
                              size_t ws_size, hipStream_t stream) {
  (void)in_sizes; (void)n_in; (void)out_size; (void)ws_size;

  const float* h_i     = (const float*)d_in[0];
  const float* eps_inf = (const float*)d_in[1];
  const float* eps_pri = (const float*)d_in[2];
  const float* Wx      = (const float*)d_in[3];
  const float* Wh      = (const float*)d_in[4];
  const float* b_lstm  = (const float*)d_in[5];
  const float* Wd[10]; const float* bd[10];
  for (int i = 1; i <= 9; ++i) {
    Wd[i] = (const float*)d_in[4 + 2 * i];
    bd[i] = (const float*)d_in[5 + 2 * i];
  }

  float* out_fake = (float*)d_out;                        // [B,T,F]
  float* out_mean = out_fake + (size_t)Bn * Tn * Fn;      // [B,T,Z]
  float* out_logv = out_mean + (size_t)Bn * Tn * Zn;      // [B,T,Z]
  float* out_z    = out_logv + (size_t)Bn * Tn * Zn;      // [T,2,B,Z]

  char* wp = (char*)d_ws;
  unsigned short* pWxh = (unsigned short*)carve(wp, (size_t)1344 * 2048 * 2);
  unsigned short* pW1  = (unsigned short*)carve(wp, (size_t)512 * 256 * 2);
  unsigned short* pW2  = (unsigned short*)carve(wp, (size_t)256 * 256 * 2);
  unsigned short* pW3  = (unsigned short*)carve(wp, (size_t)256 * 256 * 2);
  unsigned short* pW4  = (unsigned short*)carve(wp, (size_t)1536 * 64 * 2);
  unsigned short* pW5  = (unsigned short*)carve(wp, (size_t)64 * 64 * 2);
  unsigned short* pW6  = (unsigned short*)carve(wp, (size_t)64 * 64 * 2);
  unsigned short* pW7  = (unsigned short*)carve(wp, (size_t)1280 * 64 * 2);
  unsigned short* pW8  = (unsigned short*)carve(wp, (size_t)64 * 64 * 2);
  unsigned short* pW9  = (unsigned short*)carve(wp, (size_t)64 * 64 * 2);
  unsigned short* hi_bf = (unsigned short*)carve(wp, (size_t)Bn * Hn * 2);
  unsigned short* h_bf  = (unsigned short*)carve(wp, (size_t)Bn * Hn * 2);
  float*          c_f   = (float*)carve(wp, (size_t)Bn * Hn * 4);
  unsigned short* c_bf  = (unsigned short*)carve(wp, (size_t)Bn * Hn * 2);
  unsigned short* z_bf  = (unsigned short*)carve(wp, (size_t)Bn * Zn * 2);
  unsigned short* y0    = (unsigned short*)carve(wp, (size_t)Bn * Fn * 2);
  unsigned short* y1    = (unsigned short*)carve(wp, (size_t)Bn * Fn * 2);
  float*          gates = (float*)carve(wp, (size_t)Bn * 4 * Hn * 4);
  unsigned short* d1b   = (unsigned short*)carve(wp, (size_t)Bn * Fn * 2);
  unsigned short* d2b   = (unsigned short*)carve(wp, (size_t)Bn * Fn * 2);
  unsigned short* hz1b  = (unsigned short*)carve(wp, (size_t)Bn * Zn * 2);
  unsigned short* hz2b  = (unsigned short*)carve(wp, (size_t)Bn * Zn * 2);
  float*          mpf   = (float*)carve(wp, (size_t)Bn * Zn * 4);
  float*          lpf   = (float*)carve(wp, (size_t)Bn * Zn * 4);

  // --- pack all weights to WMMA-fragment tiles (bf16), once per launch ---
  pack_w_kernel<<<dim3(26, 128), 32, 0, stream>>>(Wx, pWxh, 2048, 42, 0);
  pack_w_kernel<<<dim3(16, 128), 32, 0, stream>>>(Wh, pWxh, 2048, 42, 26);
  pack_w_kernel<<<dim3(16, 16), 32, 0, stream>>>(Wd[1], pW1, 256, 16, 0);
  pack_w_kernel<<<dim3(8, 16), 32, 0, stream>>>(Wd[2], pW2, 256, 8, 0);
  pack_w_kernel<<<dim3(8, 16), 32, 0, stream>>>(Wd[3], pW3, 256, 8, 0);
  pack_w_kernel<<<dim3(48, 4), 32, 0, stream>>>(Wd[4], pW4, 64, 48, 0);
  pack_w_kernel<<<dim3(2, 4), 32, 0, stream>>>(Wd[5], pW5, 64, 2, 0);
  pack_w_kernel<<<dim3(2, 4), 32, 0, stream>>>(Wd[6], pW6, 64, 2, 0);
  pack_w_kernel<<<dim3(40, 4), 32, 0, stream>>>(Wd[7], pW7, 64, 40, 0);
  pack_w_kernel<<<dim3(2, 4), 32, 0, stream>>>(Wd[8], pW8, 64, 2, 0);
  pack_w_kernel<<<dim3(2, 4), 32, 0, stream>>>(Wd[9], pW9, 64, 2, 0);

  cvt_bf16_kernel<<<(Bn * Hn + 255) / 256, 256, 0, stream>>>(h_i, hi_bf,
                                                             Bn * Hn);
  // zero initial recurrent state (h, c, z, y_prev)
  zero_kernel<<<(Bn * Hn / 2 + 255) / 256, 256, 0, stream>>>(
      (unsigned int*)h_bf, Bn * Hn / 2);
  zero_kernel<<<(Bn * Hn + 255) / 256, 256, 0, stream>>>((unsigned int*)c_f,
                                                         Bn * Hn);
  zero_kernel<<<(Bn * Zn / 2 + 255) / 256, 256, 0, stream>>>(
      (unsigned int*)z_bf, Bn * Zn / 2);
  zero_kernel<<<(Bn * Fn / 2 + 255) / 256, 256, 0, stream>>>(
      (unsigned int*)y0, Bn * Fn / 2);

  const unsigned short* nul = nullptr;
  for (int t = 0; t < Tn; ++t) {
    unsigned short* yprev = (t & 1) ? y1 : y0;
    unsigned short* ycur  = (t & 1) ? y0 : y1;

    // gates = [z | h_i | y_prev | h] @ [Wx;Wh] + b_lstm   (K=1344, N=2048)
    launch_gemm(stream, 2048, z_bf, Zn, hi_bf, Hn, yprev, Fn, h_bf, Hn, pWxh,
                42, b_lstm, 0, gates, 2048, nullptr, 0);
    lstm_pointwise_kernel<<<(Bn * Hn + 255) / 256, 256, 0, stream>>>(
        gates, c_f, c_bf, h_bf);

    // decoder MLP -> y_j (fp32 into fake_input[:,t,:], bf16 into ycur)
    launch_gemm(stream, 256, h_bf, Hn, nul, 0, nul, 0, nul, 0, pW1, 16, bd[1],
                1, nullptr, 0, d1b, Fn);
    launch_gemm(stream, 256, d1b, Fn, nul, 0, nul, 0, nul, 0, pW2, 8, bd[2],
                1, nullptr, 0, d2b, Fn);
    launch_gemm(stream, 256, d2b, Fn, nul, 0, nul, 0, nul, 0, pW3, 8, bd[3],
                1, out_fake + t * Fn, Tn * Fn, ycur, Fn);

    // inference net: [h_i | c | y_j | y_prev] (K=1536, N=64)
    launch_gemm(stream, 64, hi_bf, Hn, c_bf, Hn, ycur, Fn, yprev, Fn, pW4, 48,
                bd[4], 1, nullptr, 0, hz1b, Zn);
    launch_gemm(stream, 64, hz1b, Zn, nul, 0, nul, 0, nul, 0, pW5, 2, bd[5],
                1, out_mean + t * Zn, Tn * Zn, nullptr, 0);
    launch_gemm(stream, 64, hz1b, Zn, nul, 0, nul, 0, nul, 0, pW6, 2, bd[6],
                1, out_logv + t * Zn, Tn * Zn, nullptr, 0);
    z_sample_kernel<<<(Bn * Zn + 255) / 256, 256, 0, stream>>>(
        out_mean + t * Zn, Tn * Zn, out_logv + t * Zn, Tn * Zn,
        eps_inf + (size_t)t * Bn * Zn, out_z + (size_t)(2 * t) * Bn * Zn,
        z_bf);

    // prior net: [h_i | c | y_prev] (K=1280, N=64)
    launch_gemm(stream, 64, hi_bf, Hn, c_bf, Hn, yprev, Fn, nul, 0, pW7, 40,
                bd[7], 1, nullptr, 0, hz2b, Zn);
    launch_gemm(stream, 64, hz2b, Zn, nul, 0, nul, 0, nul, 0, pW8, 2, bd[8],
                1, mpf, Zn, nullptr, 0);
    launch_gemm(stream, 64, hz2b, Zn, nul, 0, nul, 0, nul, 0, pW9, 2, bd[9],
                1, lpf, Zn, nullptr, 0);
    z_sample_kernel<<<(Bn * Zn + 255) / 256, 256, 0, stream>>>(
        mpf, Zn, lpf, Zn, eps_pri + (size_t)t * Bn * Zn,
        out_z + (size_t)(2 * t + 1) * Bn * Zn, nullptr);
  }
}
